// LogLinearAttentionReference_34866544509353
// MI455X (gfx1250) — compile-verified, matrix operand order checked by reading, not run on hardware
//
#include <hip/hip_runtime.h>

typedef __attribute__((ext_vector_type(16))) _Float16 v16h;
typedef __attribute__((ext_vector_type(8)))  _Float16 f16x8;
typedef __attribute__((ext_vector_type(4)))  _Float16 f16x4;
typedef __attribute__((ext_vector_type(2)))  _Float16 f16x2;
typedef __attribute__((ext_vector_type(8)))  float    v8f;

#define B_    2
#define T_    2048
#define D_    1024
#define H_    16
#define DH_   64
#define M_    (B_ * T_)   // 4096 rows of the token-major GEMMs
#define K_    1024        // reduction dim for all GEMMs
#define LAMN_ 192         // H_ * MAXLEV

// ---------------------------------------------------------------------------
// fp32 -> fp16 elementwise (x4 vectorized)
// ---------------------------------------------------------------------------
__global__ __launch_bounds__(256) void k_cvt_f16(const float4* __restrict__ in,
                                                 f16x4* __restrict__ out, int n4) {
  int i = blockIdx.x * 256 + threadIdx.x;
  if (i < n4) {
    float4 a = in[i];
    f16x4 o;
    o[0] = (_Float16)a.x; o[1] = (_Float16)a.y;
    o[2] = (_Float16)a.z; o[3] = (_Float16)a.w;
    out[i] = o;
  }
}

// ---------------------------------------------------------------------------
// transpose+convert: f32 [K,N] row-major -> f16 [N,K] row-major (K-contiguous)
// ---------------------------------------------------------------------------
__global__ __launch_bounds__(256) void k_transpose_f16(const float* __restrict__ in,
                                                       _Float16* __restrict__ out,
                                                       int K, int N) {
  __shared__ float tile[32][33];
  int bx = blockIdx.x * 32;  // n base
  int by = blockIdx.y * 32;  // k base
  int tx = threadIdx.x & 31;
  int ty = threadIdx.x >> 5; // 0..7
#pragma unroll
  for (int i = 0; i < 32; i += 8)
    tile[ty + i][tx] = in[(size_t)(by + ty + i) * N + (bx + tx)];
  __syncthreads();
#pragma unroll
  for (int i = 0; i < 32; i += 8)
    out[(size_t)(bx + ty + i) * K + (by + tx)] = (_Float16)tile[tx][ty + i];
}

// ---------------------------------------------------------------------------
// WMMA GEMM:  C[M,N] (f32) = A[M,1024](f16) * BT[N,1024](f16)^T + bias[N]
// Block tile 128(M) x (64*CTN)(N); 256 threads = 8 waves in 2(M) x 4(N).
// Each wave: 4(M) x CTN(N) 16x16 tiles, K-step 32 via v_wmma_f32_16x16x32_f16.
// K-slabs staged into double-buffered LDS with GLOBAL_LOAD_ASYNC_TO_LDS_B128
// (ASYNCcnt path), overlapped with the WMMA work on the other buffer.
// ACT=1 applies softplus (lambda head).
// ---------------------------------------------------------------------------
template <int CTN>
__global__ __launch_bounds__(256) void k_gemm_wmma(const _Float16* __restrict__ A,
                                                   const _Float16* __restrict__ BT,
                                                   const float* __restrict__ bias,
                                                   float* __restrict__ C,
                                                   int N, int ACT) {
  constexpr int BN = 64 * CTN;
  __shared__ _Float16 sA[2][128 * 32]; // [row][k] k-contiguous
  __shared__ _Float16 sB[2][BN * 32];  // [col][k] k-contiguous

  const int tid   = threadIdx.x;
  const int lane  = tid & 31;
  const int wave  = tid >> 5;
  const int waveM = wave >> 2;   // 0..1 -> 64 rows each
  const int waveN = wave & 3;    // 0..3 -> 16*CTN cols each
  const int lm    = lane & 15;   // row/col within a 16x16 tile
  const int lh    = lane >> 4;   // K-half select
  const int bm    = blockIdx.y * 128;
  const int bn    = blockIdx.x * BN;

  // async stage of one K-slab (32 halves wide) into LDS buffer `buf`
  auto stage = [&](int buf, int k0) {
#pragma unroll
    for (int i = 0; i < 2; i++) { // A: 512 x 16B chunks, 2/thread
      int idx = tid + i * 256;
      int r = idx >> 2, c = (idx & 3) * 8;
      unsigned lds = (unsigned)(unsigned long long)(void*)&sA[buf][r * 32 + c];
      unsigned long long g =
          (unsigned long long)(const void*)&A[(size_t)(bm + r) * K_ + k0 + c];
      asm volatile("global_load_async_to_lds_b128 %0, %1, off"
                   :: "v"(lds), "v"(g) : "memory");
    }
#pragma unroll
    for (int i = 0; i < CTN; i++) { // B: BN*4 x 16B chunks, CTN/thread
      int idx = tid + i * 256;
      int r = idx >> 2, c = (idx & 3) * 8;
      unsigned lds = (unsigned)(unsigned long long)(void*)&sB[buf][r * 32 + c];
      unsigned long long g =
          (unsigned long long)(const void*)&BT[(size_t)(bn + r) * K_ + k0 + c];
      asm volatile("global_load_async_to_lds_b128 %0, %1, off"
                   :: "v"(lds), "v"(g) : "memory");
    }
  };

  v8f acc[4][CTN];
#pragma unroll
  for (int i = 0; i < 4; i++)
#pragma unroll
    for (int c = 0; c < CTN; c++)
#pragma unroll
      for (int j = 0; j < 8; j++) acc[i][c][j] = 0.0f;

  stage(0, 0);
  asm volatile("s_wait_asynccnt 0x0" ::: "memory");
  __syncthreads();

  for (int k0 = 0; k0 < K_; k0 += 32) {
    const int cur = (k0 >> 5) & 1;
    const bool more = (k0 + 32) < K_;
    if (more) stage(cur ^ 1, k0 + 32); // overlap next slab with compute

    const int kb = lh * 8; // lanes 16-31 carry K halves 8..15 / 24..31
    v16h bf[CTN];
#pragma unroll
    for (int ct = 0; ct < CTN; ct++) {
      int nn = waveN * (16 * CTN) + ct * 16 + lm;
      f16x8 blo = *(const f16x8*)(&sB[cur][nn * 32 + kb]);
      f16x8 bhi = *(const f16x8*)(&sB[cur][nn * 32 + 16 + kb]);
      bf[ct] = __builtin_shufflevector(blo, bhi, 0, 1, 2, 3, 4, 5, 6, 7,
                                       8, 9, 10, 11, 12, 13, 14, 15);
    }
    v16h af[4];
#pragma unroll
    for (int rt = 0; rt < 4; rt++) {
      int m = waveM * 64 + rt * 16 + lm;
      f16x8 alo = *(const f16x8*)(&sA[cur][m * 32 + kb]);
      f16x8 ahi = *(const f16x8*)(&sA[cur][m * 32 + 16 + kb]);
      af[rt] = __builtin_shufflevector(alo, ahi, 0, 1, 2, 3, 4, 5, 6, 7,
                                       8, 9, 10, 11, 12, 13, 14, 15);
    }
#pragma unroll
    for (int rt = 0; rt < 4; rt++)
#pragma unroll
      for (int ct = 0; ct < CTN; ct++)
        acc[rt][ct] = __builtin_amdgcn_wmma_f32_16x16x32_f16(
            /*neg_a=*/false, af[rt], /*neg_b=*/false, bf[ct],
            /*c_mod=*/(short)0, acc[rt][ct],
            /*reuse_a=*/false, /*reuse_b=*/false);

    if (more) asm volatile("s_wait_asynccnt 0x0" ::: "memory");
    __syncthreads();
  }

  // epilogue: C tile layout lane l, vgpr r -> (m = 8*(l>>4)+r, n = l&15)
#pragma unroll
  for (int ct = 0; ct < CTN; ct++) {
    const int n = bn + waveN * (16 * CTN) + ct * 16 + lm;
    const float bv = bias[n];
#pragma unroll
    for (int rt = 0; rt < 4; rt++) {
#pragma unroll
      for (int r = 0; r < 8; r++) {
        int m = bm + waveM * 64 + rt * 16 + lh * 8 + r;
        float val = acc[rt][ct][r] + bv;
        if (ACT) val = (val > 20.0f) ? val : log1pf(expf(val));
        C[(size_t)m * N + n] = val;
      }
    }
  }
}

// ---------------------------------------------------------------------------
// Collapsed Fenwick scan == 2-token dyadic window:
//   y_t = lam0 * v_t * (k_t.q_t)
//       + lam1 * ( v_{t-1}(k_{t-1}.q_t) + [t even, t>=2] v_{t-2}(k_{t-2}.q_t) )
// one wave per head (DH=64 -> float2 per lane), shfl_xor dot reduction.
// ---------------------------------------------------------------------------
__global__ __launch_bounds__(512) void k_mix(const float* __restrict__ q,
                                             const float* __restrict__ k,
                                             const float* __restrict__ v,
                                             const float* __restrict__ lam,
                                             _Float16* __restrict__ y) {
  const int row  = blockIdx.x;       // b*T + t
  const int t    = row & (T_ - 1);
  const int h    = threadIdx.x >> 5; // 16 waves = 16 heads
  const int lane = threadIdx.x & 31;
  const int base = row * D_ + h * DH_ + lane * 2;

  const float2 qv = *(const float2*)(&q[base]);
  const float2 kt = *(const float2*)(&k[base]);
  const float2 vt = *(const float2*)(&v[base]);

  float p0 = qv.x * kt.x + qv.y * kt.y;
  float p1 = 0.0f, p2 = 0.0f;
  float2 v1 = make_float2(0.0f, 0.0f);
  float2 v2 = make_float2(0.0f, 0.0f);
  if (t >= 1) {
    const float2 k1 = *(const float2*)(&k[base - D_]);
    v1 = *(const float2*)(&v[base - D_]);
    p1 = qv.x * k1.x + qv.y * k1.y;
  }
  if (t >= 2 && (t & 1) == 0) {
    const float2 k2 = *(const float2*)(&k[base - 2 * D_]);
    v2 = *(const float2*)(&v[base - 2 * D_]);
    p2 = qv.x * k2.x + qv.y * k2.y;
  }
#pragma unroll
  for (int off = 16; off >= 1; off >>= 1) {
    p0 += __shfl_xor(p0, off, 32);
    p1 += __shfl_xor(p1, off, 32);
    p2 += __shfl_xor(p2, off, 32);
  }
  const float l0 = lam[row * LAMN_ + h * 12 + 0];
  const float l1 = lam[row * LAMN_ + h * 12 + 1];
  f16x2 o;
  o[0] = (_Float16)(l0 * p0 * vt.x + l1 * (p1 * v1.x + p2 * v2.x));
  o[1] = (_Float16)(l0 * p0 * vt.y + l1 * (p1 * v1.y + p2 * v2.y));
  *(f16x2*)(&y[base]) = o;
}

// ---------------------------------------------------------------------------
extern "C" void kernel_launch(void* const* d_in, const int* in_sizes, int n_in,
                              void* d_out, int out_size, void* d_ws, size_t ws_size,
                              hipStream_t stream) {
  (void)in_sizes; (void)n_in; (void)out_size; (void)ws_size;
  const float* x  = (const float*)d_in[0];
  const float* wq = (const float*)d_in[1];
  const float* bq = (const float*)d_in[2];
  const float* wk = (const float*)d_in[3];
  const float* bk = (const float*)d_in[4];
  const float* wv = (const float*)d_in[5];
  const float* bv = (const float*)d_in[6];
  const float* wl = (const float*)d_in[7];
  const float* bl = (const float*)d_in[8];
  const float* wo = (const float*)d_in[9];
  const float* bo = (const float*)d_in[10];

  // workspace carve (256B aligned)
  size_t off = 0;
  char* base = (char*)d_ws;
  auto carve = [&](size_t bytes) -> void* {
    void* p = base + off;
    off += (bytes + 255) & ~(size_t)255;
    return p;
  };
  _Float16* xh   = (_Float16*)carve((size_t)M_ * K_ * 2);
  _Float16* wqT  = (_Float16*)carve((size_t)D_ * K_ * 2);
  _Float16* wkT  = (_Float16*)carve((size_t)D_ * K_ * 2);
  _Float16* wvT  = (_Float16*)carve((size_t)D_ * K_ * 2);
  _Float16* woT  = (_Float16*)carve((size_t)D_ * K_ * 2);
  _Float16* wlT  = (_Float16*)carve((size_t)LAMN_ * K_ * 2);
  float*    qf   = (float*)carve((size_t)M_ * D_ * 4);
  float*    kf   = (float*)carve((size_t)M_ * D_ * 4);
  float*    vf   = (float*)carve((size_t)M_ * D_ * 4);
  float*    lamf = (float*)carve((size_t)M_ * LAMN_ * 4);
  _Float16* yh   = (_Float16*)carve((size_t)M_ * D_ * 2);

  // 1) convert activations to f16
  {
    int n4 = M_ * K_ / 4;
    k_cvt_f16<<<(n4 + 255) / 256, 256, 0, stream>>>((const float4*)x, (f16x4*)xh, n4);
  }
  // 2) transpose+convert weights to [N,K] f16
  k_transpose_f16<<<dim3(D_ / 32, K_ / 32), 256, 0, stream>>>(wq, wqT, K_, D_);
  k_transpose_f16<<<dim3(D_ / 32, K_ / 32), 256, 0, stream>>>(wk, wkT, K_, D_);
  k_transpose_f16<<<dim3(D_ / 32, K_ / 32), 256, 0, stream>>>(wv, wvT, K_, D_);
  k_transpose_f16<<<dim3(D_ / 32, K_ / 32), 256, 0, stream>>>(wo, woT, K_, D_);
  k_transpose_f16<<<dim3(LAMN_ / 32, K_ / 32), 256, 0, stream>>>(wl, wlT, K_, LAMN_);

  // 3) projection GEMMs (WMMA): q, k, v (128-wide tiles), lambda (64-wide, softplus)
  dim3 gqkv(D_ / 128, M_ / 128);
  k_gemm_wmma<2><<<gqkv, 256, 0, stream>>>(xh, wqT, bq, qf, D_, 0);
  k_gemm_wmma<2><<<gqkv, 256, 0, stream>>>(xh, wkT, bk, kf, D_, 0);
  k_gemm_wmma<2><<<gqkv, 256, 0, stream>>>(xh, wvT, bv, vf, D_, 0);
  k_gemm_wmma<1><<<dim3(LAMN_ / 64, M_ / 128), 256, 0, stream>>>(xh, wlT, bl, lamf, LAMN_, 1);

  // 4) collapsed scan (dyadic 2-token window)
  k_mix<<<M_, 512, 0, stream>>>(qf, kf, vf, lamf, yh);

  // 5) output projection straight into d_out (f32)
  k_gemm_wmma<2><<<gqkv, 256, 0, stream>>>(yh, woT, bo, (float*)d_out, D_, 0);
}
